// DeformableLocalCrossAttention_59863254172137
// MI455X (gfx1250) — matrix operand order repaired and checked by reading, not compiled, not run on hardware
//
#include <hip/hip_runtime.h>
#include <hip/hip_bf16.h>
#include <stdint.h>

#define BATCH 4
#define NQ    1024
#define MV    1024
#define CDIM  384
#define KNN   10
#define GRP   2
#define GC    192
#define NHEAD 6
#define HDIM  64
#define NKR   (NQ*KNN)                 // 10240
#define ROWS_MLP (BATCH*GRP*NQ*KNN)    // 81920
#define ROWS_INT (BATCH*NQ*KNN)        // 40960

typedef __attribute__((ext_vector_type(16))) __bf16 v16bf;
typedef __attribute__((ext_vector_type(8)))  float  v8f;
typedef __attribute__((ext_vector_type(4)))  unsigned int u32x4;
typedef __attribute__((ext_vector_type(8)))  int          i32x8;
typedef __attribute__((ext_vector_type(4)))  int          i32x4;

union FragB { v16bf v; uint4 q[2]; };

__device__ __forceinline__ unsigned short f2bf(float f) {
    unsigned int u = __float_as_uint(f);
    u += 0x7fffu + ((u >> 16) & 1u);       // round-to-nearest-even
    return (unsigned short)(u >> 16);
}

// ---------------------------------------------------------------- fp32 -> bf16
__global__ void cvt_bf16_kernel(const float* __restrict__ in,
                                unsigned short* __restrict__ out, int n) {
    int i = blockIdx.x * 256 + threadIdx.x;
    if (i < n) out[i] = f2bf(in[i]);
}

// ------------------------------------------------------------------ KNN top-10
__global__ __launch_bounds__(256) void knn_kernel(
    const float* __restrict__ q_pos, const float* __restrict__ v_pos,
    int* __restrict__ knn_idx, float* __restrict__ lvp) {
    __shared__ float sx[MV], sy[MV], sz[MV];
    const int b = blockIdx.x, tid = threadIdx.x;
    for (int m = tid; m < MV; m += 256) {
        const float* p = v_pos + ((size_t)b * MV + m) * 3;
        sx[m] = p[0]; sy[m] = p[1]; sz[m] = p[2];
    }
    __syncthreads();
    const int n = blockIdx.y * 256 + tid;
    const float* qp = q_pos + ((size_t)b * NQ + n) * 3;
    const float qx = qp[0], qy = qp[1], qz = qp[2];
    float bd[KNN]; int bi[KNN];
    #pragma unroll
    for (int j = 0; j < KNN; ++j) { bd[j] = 3.4e38f; bi[j] = 0; }
    for (int m = 0; m < MV; ++m) {
        float dx = qx - sx[m], dy = qy - sy[m], dz = qz - sz[m];
        float d = dx * dx + dy * dy + dz * dz;
        if (d < bd[KNN - 1]) {
            float dd = d; int ii = m;
            #pragma unroll
            for (int j = 0; j < KNN; ++j)
                if (dd < bd[j]) { float td = bd[j]; int ti = bi[j];
                                  bd[j] = dd; bi[j] = ii; dd = td; ii = ti; }
        }
    }
    const size_t base = ((size_t)b * NQ + n) * KNN;
    #pragma unroll
    for (int j = 0; j < KNN; ++j) {
        const int m = bi[j];
        knn_idx[base + j] = m;
        lvp[(base + j) * 3 + 0] = sx[m];
        lvp[(base + j) * 3 + 1] = sy[m];
        lvp[(base + j) * 3 + 2] = sz[m];
    }
}

// ----------------------------------------------- bf16 WMMA GEMM: Y = X @ W^T
// X:[R,Kd] bf16 row-major, W:[Cout,Kd] bf16 row-major, Y:[R,Cout] f32.
// Block = 256 threads (8 waves): 256 rows x 64 cols. Wave tile = 32x64
// (2 A frags x 4 B frags = 8 WMMA / K-step, B frags reused across A frags).
// 64x384 bf16 W tile staged into LDS via the Tensor Data Mover when available.
__global__ __launch_bounds__(256) void gemm_bf16_wmma_kernel(
    const unsigned short* __restrict__ X, const unsigned short* __restrict__ W,
    const float* __restrict__ bias, float* __restrict__ Y,
    int R, int Kd, int Cout) {
    __shared__ unsigned short sW[64 * 384];
    const int tid = threadIdx.x;
    const int n0 = blockIdx.y * 64;

#if defined(__gfx1250__) && __has_builtin(__builtin_amdgcn_tensor_load_to_lds) && \
    __has_builtin(__builtin_amdgcn_s_wait_tensorcnt)
    if (tid < 32) {   // wave 0 issues one 2D TDM tile load: 64 rows x 384 bf16
        const unsigned long long gaddr =
            (unsigned long long)(uintptr_t)(W + (size_t)n0 * Kd);
        const unsigned int laddr = (unsigned int)(uintptr_t)sW; // LDS byte offset
        u32x4 g0;
        g0[0] = 1u;                                  // count=1, user descriptor
        g0[1] = laddr;                               // lds_addr
        g0[2] = (unsigned int)(gaddr & 0xffffffffu); // global_addr[31:0]
        g0[3] = (unsigned int)((gaddr >> 32) & 0x1ffffffu) | (2u << 30); // type=2
        i32x8 g1;
        g1[0] = (int)(1u << 16);          // data_size = 2 bytes
        g1[1] = (int)((unsigned)Kd << 16);// tensor_dim0 = 384 (lo16)
        g1[2] = (int)(64u << 16);         // tensor_dim1 = 64 rows (lo16)
        g1[3] = (int)((unsigned)Kd << 16);// tile_dim0 = 384
        g1[4] = 64;                       // tile_dim1 = 64 (tile_dim2 = 0)
        g1[5] = Kd;                       // tensor_dim0_stride = 384
        g1[6] = 0; g1[7] = 0;             // tensor_dim1_stride unused (2D)
        i32x4 g2; g2[0] = 0; g2[1] = 0; g2[2] = 0; g2[3] = 0;
        i32x4 g3; g3[0] = 0; g3[1] = 0; g3[2] = 0; g3[3] = 0;
        i32x8 g4; g4[0] = 0; g4[1] = 0; g4[2] = 0; g4[3] = 0;
        g4[4] = 0; g4[5] = 0; g4[6] = 0; g4[7] = 0;
        __builtin_amdgcn_tensor_load_to_lds(g0, g1, g2, g3, g4, 0);
        __builtin_amdgcn_s_wait_tensorcnt(0);
    }
    __syncthreads();
#else
    {   // cooperative stage of 64x384 bf16 weight tile (48 KB)
        const uint4* src = (const uint4*)(W + (size_t)n0 * Kd);
        uint4* dst = (uint4*)sW;
        const int total = (64 * Kd) / 8;
        for (int i = tid; i < total; i += 256) dst[i] = src[i];
    }
    __syncthreads();
#endif

    const int wave = tid >> 5, lane = tid & 31;
    const int half = lane >> 4, l15 = lane & 15;
    const int r0 = (blockIdx.x * 8 + wave) * 32;

    v8f acc[2][4];
    #pragma unroll
    for (int u = 0; u < 2; ++u)
        #pragma unroll
        for (int t = 0; t < 4; ++t)
            #pragma unroll
            for (int i = 0; i < 8; ++i) acc[u][t][i] = 0.f;

    const unsigned short* aRow0 = X + (size_t)(r0 + l15) * Kd;
    const unsigned short* aRow1 = X + (size_t)(r0 + 16 + l15) * Kd;

    for (int kk = 0; kk < Kd; kk += 32) {
        __builtin_prefetch(aRow0 + kk + 64, 0, 3);
        __builtin_prefetch(aRow1 + kk + 64, 0, 3);
        FragB fa0, fa1; // A 16x32: lanes 0-15 K{0..7,16..23}, lanes 16-31 K{8..15,24..31}
        fa0.q[0] = *(const uint4*)(aRow0 + kk + half * 8);
        fa0.q[1] = *(const uint4*)(aRow0 + kk + 16 + half * 8);
        fa1.q[0] = *(const uint4*)(aRow1 + kk + half * 8);
        fa1.q[1] = *(const uint4*)(aRow1 + kk + 16 + half * 8);
        #pragma unroll
        for (int t = 0; t < 4; ++t) {
            FragB fb;  // B 32x16: lanes 0-15 K{0..7,16..23}, lanes 16-31 K{8..15,24..31}, col = W row
            const uint4* bp =
                (const uint4*)(sW + (t * 16 + l15) * Kd + kk + half * 8);
            fb.q[0] = bp[0]; fb.q[1] = bp[2];
            acc[0][t] = __builtin_amdgcn_wmma_f32_16x16x32_bf16(
                false, fa0.v, false, fb.v, (short)0, acc[0][t], false, false);
            acc[1][t] = __builtin_amdgcn_wmma_f32_16x16x32_bf16(
                false, fa1.v, false, fb.v, (short)0, acc[1][t], false, false);
        }
    }

    #pragma unroll
    for (int u = 0; u < 2; ++u)
        #pragma unroll
        for (int t = 0; t < 4; ++t) {
            const int col = n0 + t * 16 + l15;
            const float bv = bias ? bias[col] : 0.f;
            #pragma unroll
            for (int i = 0; i < 8; ++i) {
                const int row = r0 + u * 16 + half * 8 + i;
                Y[(size_t)row * Cout + col] = acc[u][t][i] + bv;
            }
        }
}

// ----------------------------------- shift_feat = [gather(v_off) | group_q] bf16
__global__ __launch_bounds__(256) void build_sfeat_kernel(
    const float* __restrict__ voff, const float* __restrict__ qp,
    const int* __restrict__ knn_idx, unsigned short* __restrict__ sfeat) {
    const int wave = threadIdx.x >> 5, lane = threadIdx.x & 31;
    const int r = blockIdx.x * 8 + wave;                // (b,g,n,k) row
    const int k = r % KNN, n = (r / KNN) % NQ;
    const int g = (r / (KNN * NQ)) % GRP, b = r / (KNN * NQ * GRP);
    const int iv = knn_idx[((size_t)b * NQ + n) * KNN + k];
    const float* sv = voff + ((size_t)b * MV + iv) * CDIM + g * GC;
    const float* sq = qp   + ((size_t)b * NQ + n ) * CDIM + g * GC;
    unsigned short* dst = sfeat + (size_t)r * CDIM;
    #pragma unroll
    for (int t = 0; t < 12; ++t) {
        const int c = lane + t * 32;
        dst[c] = f2bf(c < GC ? sv[c] : sq[c - GC]);
    }
}

// ------------------------- LayerNorm + GELU(exact) + Woff2 matvec + tanh + pos
__global__ __launch_bounds__(256) void mlp_post_kernel(
    const float* __restrict__ h, const float* __restrict__ ln_g,
    const float* __restrict__ ln_b, const float* __restrict__ Woff2,
    const float* __restrict__ lvp, float* __restrict__ shift_pos) {
    const int wave = threadIdx.x >> 5, lane = threadIdx.x & 31;
    const int r = blockIdx.x * 8 + wave;
    const float* hr = h + (size_t)r * CDIM;
    float vals[12], s = 0.f, s2 = 0.f;
    #pragma unroll
    for (int t = 0; t < 12; ++t) {
        const int c = lane + t * 32;
        const float x = hr[c]; vals[t] = x; s += x; s2 += x * x;
    }
    #pragma unroll
    for (int o = 16; o >= 1; o >>= 1) {
        s += __shfl_xor(s, o, 32); s2 += __shfl_xor(s2, o, 32);
    }
    const float mean = s * (1.f / CDIM);
    const float var  = s2 * (1.f / CDIM) - mean * mean;
    const float rstd = rsqrtf(var + 1e-5f);
    float p0 = 0.f, p1 = 0.f, p2 = 0.f;
    #pragma unroll
    for (int t = 0; t < 12; ++t) {
        const int c = lane + t * 32;
        const float x  = (vals[t] - mean) * rstd * ln_g[c] + ln_b[c];
        const float ge = 0.5f * x * (1.f + erff(x * 0.70710678118f));
        p0 += ge * Woff2[c]; p1 += ge * Woff2[CDIM + c]; p2 += ge * Woff2[2 * CDIM + c];
    }
    #pragma unroll
    for (int o = 16; o >= 1; o >>= 1) {
        p0 += __shfl_xor(p0, o, 32); p1 += __shfl_xor(p1, o, 32); p2 += __shfl_xor(p2, o, 32);
    }
    if (lane == 0) {
        const int k = r % KNN, n = (r / KNN) % NQ, b = r / (KNN * NQ * GRP);
        const size_t lp = (((size_t)b * NQ + n) * KNN + k) * 3;
        shift_pos[(size_t)r * 3 + 0] = lvp[lp + 0] + tanhf(p0);
        shift_pos[(size_t)r * 3 + 1] = lvp[lp + 1] + tanhf(p1);
        shift_pos[(size_t)r * 3 + 2] = lvp[lp + 2] + tanhf(p2);
    }
}

// --------------------------------------------------------------- three-NN top-3
__global__ __launch_bounds__(256) void three_nn_kernel(
    const float* __restrict__ shift_pos, const float* __restrict__ v_pos,
    int* __restrict__ idx3, float* __restrict__ w3) {
    __shared__ float sx[MV], sy[MV], sz[MV];
    const int bg = blockIdx.x, b = bg >> 1, tid = threadIdx.x;
    for (int m = tid; m < MV; m += 256) {
        const float* p = v_pos + ((size_t)b * MV + m) * 3;
        sx[m] = p[0]; sy[m] = p[1]; sz[m] = p[2];
    }
    __syncthreads();
    const size_t r = (size_t)bg * NKR + blockIdx.y * 256 + tid;
    const float px = shift_pos[r * 3], py = shift_pos[r * 3 + 1], pz = shift_pos[r * 3 + 2];
    float bd[3] = {3.4e38f, 3.4e38f, 3.4e38f}; int bi[3] = {0, 0, 0};
    for (int m = 0; m < MV; ++m) {
        float dx = px - sx[m], dy = py - sy[m], dz = pz - sz[m];
        float d = dx * dx + dy * dy + dz * dz;
        if (d < bd[2]) {
            float dd = d; int ii = m;
            #pragma unroll
            for (int j = 0; j < 3; ++j)
                if (dd < bd[j]) { float td = bd[j]; int ti = bi[j];
                                  bd[j] = dd; bi[j] = ii; dd = td; ii = ti; }
        }
    }
    float w[3], sw = 0.f;
    #pragma unroll
    for (int j = 0; j < 3; ++j) {
        const float dist = sqrtf(fmaxf(bd[j], 0.f));
        w[j] = 1.f / (dist + 1e-8f); sw += w[j];
    }
    const float inv = 1.f / sw;
    #pragma unroll
    for (int j = 0; j < 3; ++j) { idx3[r * 3 + j] = bi[j]; w3[r * 3 + j] = w[j] * inv; }
}

// --------------------------------- reciprocal-distance interpolation -> bf16
__global__ __launch_bounds__(256) void interp_kernel(
    const float* __restrict__ v, const int* __restrict__ idx3,
    const float* __restrict__ w3, unsigned short* __restrict__ interp) {
    const int wave = threadIdx.x >> 5, lane = threadIdx.x & 31;
    const size_t r = (size_t)blockIdx.x * 8 + wave;     // (b,g,n,k)
    const int k = (int)(r % KNN), n = (int)((r / KNN) % NQ);
    const int g = (int)((r / (KNN * NQ)) % GRP), b = (int)(r / (KNN * NQ * GRP));
    const int i0 = idx3[r * 3], i1 = idx3[r * 3 + 1], i2 = idx3[r * 3 + 2];
    const float w0 = w3[r * 3], w1 = w3[r * 3 + 1], w2 = w3[r * 3 + 2];
    const float* v0 = v + ((size_t)b * MV + i0) * CDIM + g * GC;
    const float* v1 = v + ((size_t)b * MV + i1) * CDIM + g * GC;
    const float* v2 = v + ((size_t)b * MV + i2) * CDIM + g * GC;
    unsigned short* dst = interp + (((size_t)b * NQ + n) * KNN + k) * CDIM + g * GC;
    #pragma unroll
    for (int t = 0; t < 6; ++t) {
        const int c = lane + t * 32;
        dst[c] = f2bf(w0 * v0[c] + w1 * v1[c] + w2 * v2[c]);
    }
}

// ------------------------------------- local attention over K=10 per (b,n,h)
__global__ __launch_bounds__(256) void attn_kernel(
    const float* __restrict__ qp, const float* __restrict__ kf,
    const float* __restrict__ vf, unsigned short* __restrict__ outbf) {
    const int idx = blockIdx.x * 256 + threadIdx.x;     // 24576 threads
    const int h = idx % NHEAD, n = (idx / NHEAD) % NQ, b = idx / (NHEAD * NQ);
    const float* qrow = qp + ((size_t)b * NQ + n) * CDIM + h * HDIM;
    float qv[HDIM];
    #pragma unroll
    for (int d = 0; d < HDIM; ++d) qv[d] = qrow[d];
    float a[KNN], mx = -3.4e38f;
    for (int k = 0; k < KNN; ++k) {
        const float* kr = kf + (((size_t)b * NQ + n) * KNN + k) * CDIM + h * HDIM;
        float s = 0.f;
        #pragma unroll
        for (int d = 0; d < HDIM; ++d) s += qv[d] * kr[d];
        s *= 0.125f;                                     // HD^-0.5
        a[k] = s; mx = fmaxf(mx, s);
    }
    float se = 0.f;
    #pragma unroll
    for (int k = 0; k < KNN; ++k) { a[k] = expf(a[k] - mx); se += a[k]; }
    const float inv = 1.f / se;
    float o[HDIM];
    #pragma unroll
    for (int d = 0; d < HDIM; ++d) o[d] = 0.f;
    for (int k = 0; k < KNN; ++k) {
        const float w = a[k] * inv;
        const float* vr = vf + (((size_t)b * NQ + n) * KNN + k) * CDIM + h * HDIM;
        #pragma unroll
        for (int d = 0; d < HDIM; ++d) o[d] += w * vr[d];
    }
    unsigned short* dst = outbf + ((size_t)b * NQ + n) * CDIM + h * HDIM;
    #pragma unroll
    for (int d = 0; d < HDIM; ++d) dst[d] = f2bf(o[d]);
}

// =============================================================================
extern "C" void kernel_launch(void* const* d_in, const int* in_sizes, int n_in,
                              void* d_out, int out_size, void* d_ws, size_t ws_size,
                              hipStream_t stream) {
    (void)in_sizes; (void)n_in; (void)out_size; (void)ws_size;
    const float* q     = (const float*)d_in[0];
    const float* q_pos = (const float*)d_in[1];
    const float* v     = (const float*)d_in[2];
    const float* v_pos = (const float*)d_in[3];
    const float* Wq    = (const float*)d_in[4];
    const float* Wk    = (const float*)d_in[5];
    const float* Wv    = (const float*)d_in[6];
    const float* Wvoff = (const float*)d_in[7];
    const float* Woff1 = (const float*)d_in[8];
    const float* boff1 = (const float*)d_in[9];
    const float* ln_g  = (const float*)d_in[10];
    const float* ln_b  = (const float*)d_in[11];
    const float* Woff2 = (const float*)d_in[12];
    const float* Wproj = (const float*)d_in[13];
    const float* bproj = (const float*)d_in[14];
    float* out = (float*)d_out;

    char* ws = (char*)d_ws;
    size_t off = 0;
    auto alloc = [&](size_t bytes) -> void* {
        void* p = ws + off;
        off = (off + bytes + 255) & ~(size_t)255;
        return p;
    };
    const size_t WN = (size_t)CDIM * CDIM;                 // 147456
    const size_t QN = (size_t)BATCH * NQ * CDIM;           // 1572864

    unsigned short* Wq_bf    = (unsigned short*)alloc(WN * 2);
    unsigned short* Wk_bf    = (unsigned short*)alloc(WN * 2);
    unsigned short* Wv_bf    = (unsigned short*)alloc(WN * 2);
    unsigned short* Wvoff_bf = (unsigned short*)alloc(WN * 2);
    unsigned short* Woff1_bf = (unsigned short*)alloc(WN * 2);
    unsigned short* Wproj_bf = (unsigned short*)alloc(WN * 2);
    unsigned short* q_bf     = (unsigned short*)alloc(QN * 2);
    unsigned short* v_bf     = (unsigned short*)alloc(QN * 2);
    int*   knn_idx  = (int*)  alloc((size_t)ROWS_INT * 4);
    float* lvp      = (float*)alloc((size_t)ROWS_INT * 3 * 4);
    float* qp_f     = (float*)alloc(QN * 4);
    float* voff_f   = (float*)alloc(QN * 4);
    unsigned short* sfeat_bf = (unsigned short*)alloc((size_t)ROWS_MLP * CDIM * 2);
    float* h_f      = (float*)alloc((size_t)ROWS_MLP * CDIM * 4);
    float* shiftp   = (float*)alloc((size_t)ROWS_MLP * 3 * 4);
    int*   idx3     = (int*)  alloc((size_t)ROWS_MLP * 3 * 4);
    float* w3       = (float*)alloc((size_t)ROWS_MLP * 3 * 4);
    unsigned short* interp_bf = (unsigned short*)alloc((size_t)ROWS_INT * CDIM * 2);
    float* kf_f     = (float*)alloc((size_t)ROWS_INT * CDIM * 4);
    float* vf_f     = (float*)alloc((size_t)ROWS_INT * CDIM * 4);
    unsigned short* ao_bf = (unsigned short*)alloc(QN * 2);

    // 1) fp32 -> bf16 conversions
    cvt_bf16_kernel<<<(int)((WN + 255) / 256), 256, 0, stream>>>(Wq,    Wq_bf,    (int)WN);
    cvt_bf16_kernel<<<(int)((WN + 255) / 256), 256, 0, stream>>>(Wk,    Wk_bf,    (int)WN);
    cvt_bf16_kernel<<<(int)((WN + 255) / 256), 256, 0, stream>>>(Wv,    Wv_bf,    (int)WN);
    cvt_bf16_kernel<<<(int)((WN + 255) / 256), 256, 0, stream>>>(Wvoff, Wvoff_bf, (int)WN);
    cvt_bf16_kernel<<<(int)((WN + 255) / 256), 256, 0, stream>>>(Woff1, Woff1_bf, (int)WN);
    cvt_bf16_kernel<<<(int)((WN + 255) / 256), 256, 0, stream>>>(Wproj, Wproj_bf, (int)WN);
    cvt_bf16_kernel<<<(int)((QN + 255) / 256), 256, 0, stream>>>(q, q_bf, (int)QN);
    cvt_bf16_kernel<<<(int)((QN + 255) / 256), 256, 0, stream>>>(v, v_bf, (int)QN);

    // 2) KNN (top-10) + local_v_pos gather
    knn_kernel<<<dim3(BATCH, NQ / 256), 256, 0, stream>>>(q_pos, v_pos, knn_idx, lvp);

    // 3) qp = q @ Wq^T ; v_off = v @ Wvoff^T     (R = 4096 -> 16 row-blocks)
    gemm_bf16_wmma_kernel<<<dim3(16, 6), 256, 0, stream>>>(q_bf, Wq_bf,    nullptr, qp_f,   BATCH * NQ, CDIM, CDIM);
    gemm_bf16_wmma_kernel<<<dim3(16, 6), 256, 0, stream>>>(v_bf, Wvoff_bf, nullptr, voff_f, BATCH * MV, CDIM, CDIM);

    // 4) shift_feat (bf16), then h = shift_feat @ Woff1^T + boff1   (R = 81920)
    build_sfeat_kernel<<<ROWS_MLP / 8, 256, 0, stream>>>(voff_f, qp_f, knn_idx, sfeat_bf);
    gemm_bf16_wmma_kernel<<<dim3(ROWS_MLP / 256, 6), 256, 0, stream>>>(sfeat_bf, Woff1_bf, boff1, h_f, ROWS_MLP, CDIM, CDIM);

    // 5) LN + GELU + Woff2 + tanh -> shift_pos
    mlp_post_kernel<<<ROWS_MLP / 8, 256, 0, stream>>>(h_f, ln_g, ln_b, Woff2, lvp, shiftp);

    // 6) three-NN + weights, then interpolation -> interp (bf16)
    three_nn_kernel<<<dim3(BATCH * GRP, NKR / 256), 256, 0, stream>>>(shiftp, v_pos, idx3, w3);
    interp_kernel<<<ROWS_MLP / 8, 256, 0, stream>>>(v, idx3, w3, interp_bf);

    // 7) kf / vf projections    (R = 40960)
    gemm_bf16_wmma_kernel<<<dim3(ROWS_INT / 256, 6), 256, 0, stream>>>(interp_bf, Wk_bf, nullptr, kf_f, ROWS_INT, CDIM, CDIM);
    gemm_bf16_wmma_kernel<<<dim3(ROWS_INT / 256, 6), 256, 0, stream>>>(interp_bf, Wv_bf, nullptr, vf_f, ROWS_INT, CDIM, CDIM);

    // 8) local cross-attention -> bf16 rows
    attn_kernel<<<(BATCH * NQ * NHEAD) / 256, 256, 0, stream>>>(qp_f, kf_f, vf_f, ao_bf);

    // 9) out = attn_out @ Wproj^T + bproj        (R = 4096, fp32 to d_out)
    gemm_bf16_wmma_kernel<<<dim3(16, 6), 256, 0, stream>>>(ao_bf, Wproj_bf, bproj, out, BATCH * NQ, CDIM, CDIM);
}
